// ViTBackbone_15066745274462
// MI455X (gfx1250) — matrix operand order repaired and checked by reading, not compiled, not run on hardware
//
#include <hip/hip_runtime.h>
#include <hip/hip_fp16.h>

// ---------------------------------------------------------------------------
// Types for CDNA5 WMMA (wave32): 16x16x32 f16 -> f32
// ---------------------------------------------------------------------------
typedef __attribute__((ext_vector_type(16))) _Float16 v16h;
typedef __attribute__((ext_vector_type(8)))  float    v8f;

union FragH { v16h v; uint4 q[2]; };

__device__ __forceinline__ v8f wmma16(const FragH& a, const FragH& b, v8f c) {
  // (neg_a, A, neg_b, B, c_mod, C, reuse_a, reuse_b)
  return __builtin_amdgcn_wmma_f32_16x16x32_f16(false, a.v, false, b.v,
                                                (short)0, c, false, false);
}

// ---------------------------------------------------------------------------
// Problem constants
// ---------------------------------------------------------------------------
constexpr int kB   = 32;
constexpr int kC   = 3;
constexpr int kIMG = 224;
constexpr int kP   = 16;
constexpr int kE   = 768;
constexpr int kNH  = 12;
constexpr int kD   = 64;
constexpr int kM   = 3072;
constexpr int kL   = 12;
constexpr int kN   = 197;          // tokens
constexpr int kNP  = 224;          // padded tokens (multiple of 32)
constexpr int kT   = kB * kN;      // 6304 rows (= 197*32, multiple of 32)
constexpr int kBH  = kB * kNH;     // 384
constexpr int kHP  = kIMG / kP;    // 14
constexpr int kNPAT= kHP * kHP;    // 196 patches
constexpr int kTP  = kB * kNPAT;   // 6272 patch rows (= 196*32)

// ---------------------------------------------------------------------------
// fp32 -> f16 elementwise (conv_w is already [N][K] = [E][C*P*P])
// ---------------------------------------------------------------------------
__global__ void k_cvt(const float* __restrict__ w, _Float16* __restrict__ o,
                      size_t n) {
  size_t i = (size_t)blockIdx.x * 256 + threadIdx.x;
  if (i < n) o[i] = (_Float16)w[i];
}

// ---------------------------------------------------------------------------
// Tiled transpose-convert: W[K,N] fp32 -> WT[N,K] f16 (coalesced both ways)
// grid (K/32, N/32), 256 threads
// ---------------------------------------------------------------------------
__global__ __launch_bounds__(256)
void k_tcvt(const float* __restrict__ w, _Float16* __restrict__ wt,
            int K, int N) {
  __shared__ float tile[32][33];
  const int k0 = blockIdx.x * 32;
  const int n0 = blockIdx.y * 32;
  const int tx = threadIdx.x & 31;
  const int ty = threadIdx.x >> 5;          // 0..7
#pragma unroll
  for (int j = 0; j < 32; j += 8)
    tile[ty + j][tx] = w[(size_t)(k0 + ty + j) * N + n0 + tx];
  __syncthreads();
#pragma unroll
  for (int j = 0; j < 32; j += 8)
    wt[(size_t)(n0 + ty + j) * K + k0 + tx] = (_Float16)tile[tx][ty + j];
}

// ---------------------------------------------------------------------------
// Patchify: x[B,C,224,224] -> patches f16 [B*196, 768] (c*256 + iy*16 + ix)
// ---------------------------------------------------------------------------
__global__ void k_patchify(const float* __restrict__ x,
                           _Float16* __restrict__ p) {
  size_t i = (size_t)blockIdx.x * 256 + threadIdx.x;
  if (i >= (size_t)kTP * kE) return;
  int col  = (int)(i % kE);
  int rowp = (int)(i / kE);
  int b  = rowp / kNPAT, pp = rowp % kNPAT;
  int py = pp / kHP,     px = pp % kHP;
  int c  = col >> 8, iy = (col >> 4) & 15, ix = col & 15;
  p[i] = (_Float16)x[(((size_t)b * kC + c) * kIMG + (py * kP + iy)) * kIMG
                     + px * kP + ix];
}

// ---------------------------------------------------------------------------
// Fragment-direct WMMA GEMM:  C[M,N] = A[M,K](f16) * BT[N,K](f16)^T + bias[N]
// block = 128 threads (4 waves); each wave computes 32 rows x 64 cols
// (2 A-frags, 8 f32 accumulators) so each B fragment feeds 2 WMMAs from
// registers; the 4 waves share n0 so the WGP$ catches inter-wave B reuse.
// M multiple of 32; N of 64; K of 32.
// ---------------------------------------------------------------------------
template <bool OUT16, bool GELU>
__global__ __launch_bounds__(128)
void k_gemm(const _Float16* __restrict__ A, const _Float16* __restrict__ BT,
            const float* __restrict__ bias, float* __restrict__ C32,
            _Float16* __restrict__ C16, int M, int N, int K) {
  const int lane = threadIdx.x & 31;
  const int wave = threadIdx.x >> 5;
  const int m0   = (blockIdx.x * 4 + wave) * 32;
  const int n0   = blockIdx.y * 64;
  if (m0 >= M) return;                 // wave-uniform: EXEC stays all-ones
  const int lm = lane & 15;
  const int lg = lane >> 4;

  const _Float16* arow0 = A + (size_t)(m0 + lm) * K;
  const _Float16* arow1 = A + (size_t)(m0 + 16 + lm) * K;

  v8f acc[2][4];
#pragma unroll
  for (int t = 0; t < 4; ++t) {
    float bv = bias[n0 + t * 16 + lm];
    acc[0][t] = (v8f){bv, bv, bv, bv, bv, bv, bv, bv};
    acc[1][t] = acc[0][t];
  }

  for (int kb = 0; kb < K; kb += 32) {
    FragH a0, a1;                                 // two 16x32 A frags
    a0.q[0] = *reinterpret_cast<const uint4*>(arow0 + kb + 16 * lg);
    a0.q[1] = *reinterpret_cast<const uint4*>(arow0 + kb + 16 * lg + 8);
    a1.q[0] = *reinterpret_cast<const uint4*>(arow1 + kb + 16 * lg);
    a1.q[1] = *reinterpret_cast<const uint4*>(arow1 + kb + 16 * lg + 8);
    __builtin_prefetch(arow0 + kb + 128, 0, 1);   // global_prefetch_b8
    __builtin_prefetch(arow1 + kb + 128, 0, 1);
#pragma unroll
    for (int t = 0; t < 4; ++t) {
      const _Float16* brow =
          BT + (size_t)(n0 + t * 16 + lm) * K + kb + 16 * lg;
      FragH b;                                    // B frag: 32x16 (K-contig)
      b.q[0] = reinterpret_cast<const uint4*>(brow)[0];
      b.q[1] = reinterpret_cast<const uint4*>(brow)[1];
      acc[0][t] = wmma16(a0, b, acc[0][t]);
      acc[1][t] = wmma16(a1, b, acc[1][t]);
    }
  }

#pragma unroll
  for (int r = 0; r < 2; ++r) {
#pragma unroll
    for (int t = 0; t < 4; ++t) {
#pragma unroll
      for (int i = 0; i < 8; ++i) {
        int rr = m0 + r * 16 + i + 8 * lg;        // C layout: M = i + 8*(l/16)
        int cc = n0 + t * 16 + lm;
        float v = acc[r][t][i];
        if (GELU) v = 0.5f * v * (1.0f + erff(v * 0.70710678118654752f));
        if (OUT16) C16[(size_t)rr * N + cc] = (_Float16)v;
        else       C32[(size_t)rr * N + cc] = v;
      }
    }
  }
}

// ---------------------------------------------------------------------------
// Assemble tokens: h = concat(cls, patch_embed) + pos ; fp32 + f16 mirror
// ---------------------------------------------------------------------------
__global__ __launch_bounds__(256)
void k_assemble(const float* __restrict__ pe, const float* __restrict__ cls,
                const float* __restrict__ pos, float* __restrict__ h32,
                _Float16* __restrict__ h16) {
  int row = blockIdx.x;                 // 0..6303
  int b = row / kN, n = row % kN;
  for (int e = threadIdx.x; e < kE; e += 256) {
    float v = (n == 0) ? cls[e] : pe[(size_t)(b * kNPAT + n - 1) * kE + e];
    v += pos[(size_t)n * kE + e];
    h32[(size_t)row * kE + e] = v;
    h16[(size_t)row * kE + e] = (_Float16)v;
  }
}

// ---------------------------------------------------------------------------
// QKV split + per-head LayerNorm on q,k.  One block (768 threads) per token.
//   qh/kh : [BH, 224, 64] f16 (padded rows pre-zeroed)
//   vT    : [BH, 64, 224] f16 (transposed, padded cols pre-zeroed)
// ---------------------------------------------------------------------------
__global__ __launch_bounds__(768)
void k_qkv_split_ln(const float* __restrict__ qkv, const float* __restrict__ qg,
                    const float* __restrict__ qb, const float* __restrict__ kg,
                    const float* __restrict__ kbias,
                    _Float16* __restrict__ qh, _Float16* __restrict__ kh,
                    _Float16* __restrict__ vT) {
  const int row = blockIdx.x;           // token row
  const int b = row / kN, n = row % kN;
  const int e = threadIdx.x;            // 0..767
  const int h = e >> 6, d = e & 63;
  const float* base = qkv + (size_t)row * (3 * kE);
  float q = base[e], k = base[kE + e], v = base[2 * kE + e];

  __shared__ float red[768];
  auto ln64 = [&](float xv, const float* g, const float* bb) -> float {
    red[e] = xv; __syncthreads();
#pragma unroll
    for (int s = 32; s; s >>= 1) {
      if (d < s) red[e] += red[e + s];
      __syncthreads();
    }
    float m = red[h * 64] * (1.0f / 64.0f); __syncthreads();
    float dx = xv - m;
    red[e] = dx * dx; __syncthreads();
#pragma unroll
    for (int s = 32; s; s >>= 1) {
      if (d < s) red[e] += red[e + s];
      __syncthreads();
    }
    float var = red[h * 64] * (1.0f / 64.0f); __syncthreads();
    return dx * rsqrtf(var + 1e-5f) * g[d] + bb[d];
  };

  float qn = ln64(q, qg, qb);
  float kn = ln64(k, kg, kbias);

  size_t bh = (size_t)(b * kNH + h);
  qh[(bh * kNP + n) * kD + d] = (_Float16)qn;
  kh[(bh * kNP + n) * kD + d] = (_Float16)kn;
  vT[(bh * kD + d) * kNP + n] = (_Float16)v;
}

// ---------------------------------------------------------------------------
// Fused attention: one wave per (bh, 16-query tile).
//   S = Q K^T / 8 -> LDS; masked softmax; O = P V / rowsum -> o16[B,N,H*64]
// ---------------------------------------------------------------------------
__global__ __launch_bounds__(32)
void k_attn(const _Float16* __restrict__ qh, const _Float16* __restrict__ kh,
            const _Float16* __restrict__ vT, _Float16* __restrict__ o16) {
  const int bh = blockIdx.y;
  const int q0 = blockIdx.x * 16;
  const int lane = threadIdx.x;
  const int lm = lane & 15, lg = lane >> 4;

  __shared__ float S[16][kNP];
  __shared__ __align__(16) _Float16 Pb[16][kNP];
  __shared__ float rsum[16];

  const _Float16* qbase = qh + (size_t)bh * kNP * kD;
  const _Float16* kbase = kh + (size_t)bh * kNP * kD;
  const _Float16* vbase = vT + (size_t)bh * kD * kNP;

  // Q fragments: 16 rows x 64 (two 16x32 A-frags)
  FragH aq[2];
  const _Float16* qr = qbase + (size_t)(q0 + lm) * kD;
  aq[0].q[0] = *reinterpret_cast<const uint4*>(qr + 16 * lg);
  aq[0].q[1] = *reinterpret_cast<const uint4*>(qr + 16 * lg + 8);
  aq[1].q[0] = *reinterpret_cast<const uint4*>(qr + 32 + 16 * lg);
  aq[1].q[1] = *reinterpret_cast<const uint4*>(qr + 32 + 16 * lg + 8);

  // S = Q K^T (14 key tiles of 16)
  for (int kt = 0; kt < kNP / 16; ++kt) {
    const _Float16* kr = kbase + (size_t)(kt * 16 + lm) * kD;
    FragH b0, b1;
    b0.q[0] = *reinterpret_cast<const uint4*>(kr + 16 * lg);
    b0.q[1] = *reinterpret_cast<const uint4*>(kr + 16 * lg + 8);
    b1.q[0] = *reinterpret_cast<const uint4*>(kr + 32 + 16 * lg);
    b1.q[1] = *reinterpret_cast<const uint4*>(kr + 32 + 16 * lg + 8);
    v8f s = (v8f){0, 0, 0, 0, 0, 0, 0, 0};
    s = wmma16(aq[0], b0, s);
    s = wmma16(aq[1], b1, s);
#pragma unroll
    for (int i = 0; i < 8; ++i)
      S[i + 8 * lg][kt * 16 + lm] = s[i] * 0.125f;   // 1/sqrt(64)
  }
  __syncthreads();

  // masked softmax per row (224 cols, keys >=197 masked)
  for (int r = 0; r < 16; ++r) {
    float mx = -1e30f;
    for (int c = lane; c < kNP; c += 32) {
      float x = (c < kN) ? S[r][c] : -1e30f;
      S[r][c] = x;
      mx = fmaxf(mx, x);
    }
#pragma unroll
    for (int o = 16; o; o >>= 1) mx = fmaxf(mx, __shfl_xor(mx, o, 32));
    float sm = 0.0f;
    for (int c = lane; c < kNP; c += 32) {
      float ev = __expf(S[r][c] - mx);
      Pb[r][c] = (_Float16)ev;
      sm += ev;
    }
#pragma unroll
    for (int o = 16; o; o >>= 1) sm += __shfl_xor(sm, o, 32);
    if (lane == 0) rsum[r] = sm;
  }
  __syncthreads();

  // O = P V (7 key chunks of 32, 4 d-tiles of 16)
  v8f acc[4];
#pragma unroll
  for (int t = 0; t < 4; ++t) acc[t] = (v8f){0, 0, 0, 0, 0, 0, 0, 0};
  const _Float16* prow = &Pb[lm][0];
  for (int kc = 0; kc < kNP; kc += 32) {
    FragH ap;
    ap.q[0] = *reinterpret_cast<const uint4*>(prow + kc + 16 * lg);
    ap.q[1] = *reinterpret_cast<const uint4*>(prow + kc + 16 * lg + 8);
#pragma unroll
    for (int t = 0; t < 4; ++t) {
      const _Float16* vr = vbase + (size_t)(t * 16 + lm) * kNP + kc + 16 * lg;
      FragH bv;
      bv.q[0] = reinterpret_cast<const uint4*>(vr)[0];
      bv.q[1] = reinterpret_cast<const uint4*>(vr)[1];
      acc[t] = wmma16(ap, bv, acc[t]);
    }
  }

  const int b = bh / kNH, h = bh % kNH;
#pragma unroll
  for (int t = 0; t < 4; ++t) {
#pragma unroll
    for (int i = 0; i < 8; ++i) {
      int qrow = q0 + i + 8 * lg;
      if (qrow < kN) {
        float ov = acc[t][i] / rsum[i + 8 * lg];
        o16[(size_t)(b * kN + qrow) * kE + h * kD + t * 16 + lm] = (_Float16)ov;
      }
    }
  }
}

// ---------------------------------------------------------------------------
// h_new = LayerNorm(x)*g + b + h_res (in place on h32, plus f16 mirror)
// ---------------------------------------------------------------------------
__global__ __launch_bounds__(256)
void k_ln_add(const float* __restrict__ x, const float* __restrict__ g,
              const float* __restrict__ bb, float* __restrict__ h32,
              _Float16* __restrict__ h16) {
  const int row = blockIdx.x;
  const int tid = threadIdx.x;
  const float* xr = x + (size_t)row * kE;
  __shared__ float red[256];

  float s = 0.0f;
  for (int e = tid; e < kE; e += 256) s += xr[e];
  red[tid] = s; __syncthreads();
#pragma unroll
  for (int t = 128; t; t >>= 1) {
    if (tid < t) red[tid] += red[tid + t];
    __syncthreads();
  }
  float mean = red[0] * (1.0f / kE); __syncthreads();

  float s2 = 0.0f;
  for (int e = tid; e < kE; e += 256) {
    float d = xr[e] - mean;
    s2 += d * d;
  }
  red[tid] = s2; __syncthreads();
#pragma unroll
  for (int t = 128; t; t >>= 1) {
    if (tid < t) red[tid] += red[tid + t];
    __syncthreads();
  }
  float rstd = rsqrtf(red[0] * (1.0f / kE) + 1e-5f);

  for (int e = tid; e < kE; e += 256) {
    float y = (xr[e] - mean) * rstd * g[e] + bb[e] + h32[(size_t)row * kE + e];
    h32[(size_t)row * kE + e] = y;
    h16[(size_t)row * kE + e] = (_Float16)y;
  }
}

__global__ void k_extract(const float* __restrict__ h32, float* __restrict__ out) {
  int i = blockIdx.x * 256 + threadIdx.x;
  if (i < kB * kE) {
    int b = i / kE, e = i % kE;
    out[i] = h32[(size_t)b * kN * kE + e];       // token 0 of each batch
  }
}

// ---------------------------------------------------------------------------
// Host orchestration
// ---------------------------------------------------------------------------
static inline int ceil_div(long long a, long long b) { return (int)((a + b - 1) / b); }

extern "C" void kernel_launch(void* const* d_in, const int* in_sizes, int n_in,
                              void* d_out, int out_size, void* d_ws, size_t ws_size,
                              hipStream_t stream) {
  const float* x      = (const float*)d_in[0];
  const float* conv_w = (const float*)d_in[1];
  const float* conv_b = (const float*)d_in[2];
  const float* cls_t  = (const float*)d_in[3];
  const float* pos    = (const float*)d_in[4];
  const float* qkv_w  = (const float*)d_in[5];
  const float* qkv_b  = (const float*)d_in[6];
  const float* qn_g   = (const float*)d_in[7];
  const float* qn_b   = (const float*)d_in[8];
  const float* kn_g   = (const float*)d_in[9];
  const float* kn_b   = (const float*)d_in[10];
  const float* proj_w = (const float*)d_in[11];
  const float* proj_b = (const float*)d_in[12];
  const float* ln1_g  = (const float*)d_in[13];
  const float* ln1_b  = (const float*)d_in[14];
  const float* w1     = (const float*)d_in[15];
  const float* b1     = (const float*)d_in[16];
  const float* w2     = (const float*)d_in[17];
  const float* b2     = (const float*)d_in[18];
  const float* ln2_g  = (const float*)d_in[19];
  const float* ln2_b  = (const float*)d_in[20];

  char* ws = (char*)d_ws;
  size_t off = 0;
  auto take = [&](size_t bytes) -> char* {
    char* p = ws + off;
    off = (off + bytes + 255) & ~(size_t)255;
    return p;
  };

  float*    hf32  = (float*)take((size_t)kT * kE * 4);
  _Float16* hf16  = (_Float16*)take((size_t)kT * kE * 2);
  _Float16* w16   = (_Float16*)take((size_t)kM * kE * 2);       // reused per GEMM
  float*    qkv32 = (float*)take((size_t)kT * 3 * kE * 4);      // also mlp16
  _Float16* mlp16 = (_Float16*)qkv32;
  _Float16* qh16  = (_Float16*)take((size_t)kBH * kNP * kD * 2);
  _Float16* kh16  = (_Float16*)take((size_t)kBH * kNP * kD * 2);
  _Float16* vT16  = (_Float16*)take((size_t)kBH * kD * kNP * 2);
  float*    x32a  = (float*)take((size_t)kT * kE * 4);
  _Float16* of16  = (_Float16*)take((size_t)kT * kE * 2);       // also patches

  // zero padded attention buffers once (pads stay zero across layers)
  hipMemsetAsync(qh16, 0, (size_t)kBH * kNP * kD * 2, stream);
  hipMemsetAsync(kh16, 0, (size_t)kBH * kNP * kD * 2, stream);
  hipMemsetAsync(vT16, 0, (size_t)kBH * kD * kNP * 2, stream);

  // ---- patch embedding: patches @ conv_w^T + conv_b ----
  k_patchify<<<ceil_div((long long)kTP * kE, 256), 256, 0, stream>>>(x, of16);
  k_cvt<<<ceil_div((long long)kE * kE, 256), 256, 0, stream>>>(conv_w, w16,
                                                               (size_t)kE * kE);
  {
    dim3 g(ceil_div(kTP / 32, 4), kE / 64);
    k_gemm<false, false><<<g, 128, 0, stream>>>(of16, w16, conv_b, x32a,
                                                nullptr, kTP, kE, kE);
  }
  k_assemble<<<kT, 256, 0, stream>>>(x32a, cls_t, pos, hf32, hf16);

  const dim3 gQKV(ceil_div(kT / 32, 4), (3 * kE) / 64);
  const dim3 gE  (ceil_div(kT / 32, 4), kE / 64);
  const dim3 gM  (ceil_div(kT / 32, 4), kM / 64);

  for (int l = 0; l < kL; ++l) {
    // QKV projection
    k_tcvt<<<dim3(kE / 32, (3 * kE) / 32), 256, 0, stream>>>(
        qkv_w + (size_t)l * kE * 3 * kE, w16, kE, 3 * kE);
    k_gemm<false, false><<<gQKV, 128, 0, stream>>>(
        hf16, w16, qkv_b + (size_t)l * 3 * kE, qkv32, nullptr, kT, 3 * kE, kE);

    // split heads + q/k LayerNorm
    k_qkv_split_ln<<<kT, 768, 0, stream>>>(
        qkv32, qn_g + (size_t)l * kD, qn_b + (size_t)l * kD,
        kn_g + (size_t)l * kD, kn_b + (size_t)l * kD, qh16, kh16, vT16);

    // fused attention
    k_attn<<<dim3(ceil_div(kN, 16), kBH), 32, 0, stream>>>(qh16, kh16, vT16,
                                                           of16);

    // output projection + LN + residual
    k_tcvt<<<dim3(kE / 32, kE / 32), 256, 0, stream>>>(
        proj_w + (size_t)l * kE * kE, w16, kE, kE);
    k_gemm<false, false><<<gE, 128, 0, stream>>>(
        of16, w16, proj_b + (size_t)l * kE, x32a, nullptr, kT, kE, kE);
    k_ln_add<<<kT, 256, 0, stream>>>(x32a, ln1_g + (size_t)l * kE,
                                     ln1_b + (size_t)l * kE, hf32, hf16);

    // MLP: GELU(h@W1+b1)@W2+b2, then LN + residual
    k_tcvt<<<dim3(kE / 32, kM / 32), 256, 0, stream>>>(
        w1 + (size_t)l * kE * kM, w16, kE, kM);
    k_gemm<true, true><<<gM, 128, 0, stream>>>(
        hf16, w16, b1 + (size_t)l * kM, nullptr, mlp16, kT, kM, kE);
    k_tcvt<<<dim3(kM / 32, kE / 32), 256, 0, stream>>>(
        w2 + (size_t)l * kM * kE, w16, kM, kE);
    k_gemm<false, false><<<gE, 128, 0, stream>>>(
        mlp16, w16, b2 + (size_t)l * kE, x32a, nullptr, kT, kE, kM);
    k_ln_add<<<kT, 256, 0, stream>>>(x32a, ln2_g + (size_t)l * kE,
                                     ln2_b + (size_t)l * kE, hf32, hf16);
  }

  k_extract<<<ceil_div(kB * kE, 256), 256, 0, stream>>>(hf32, (float*)d_out);
}
